// YOLOv6Head_39814346834356
// MI455X (gfx1250) — matrix operand order repaired and checked by reading, not compiled, not run on hardware
//
#include <hip/hip_runtime.h>
#include <cstdint>
#include <cstddef>

// ---------------------------------------------------------------------------
// YOLOv6/SimOTA loss for MI455X (gfx1250, wave32).
// cls-cost computed as onehot(G x NC) @ M(NC x A) via v_wmma_f32_16x16x32_f16.
// ---------------------------------------------------------------------------

typedef __attribute__((ext_vector_type(16))) _Float16 v16h;
typedef __attribute__((ext_vector_type(8)))  float    v8f;

constexpr int A_  = 8400;   // total anchors: 6400 + 1600 + 400
constexpr int B_  = 16;
constexpr int G_  = 60;
constexpr int NC_ = 80;
constexpr int GP_ = 64;     // padded G rows for cost matrix (4 x 16)
constexpr float PI_F = 3.14159265358979323846f;

__device__ __forceinline__ float sigm(float x) { return 1.0f / (1.0f + expf(-x)); }

__device__ __forceinline__ float flat_ld(const float* o0, const float* o1, const float* o2,
                                         int b, int a, int c) {
  if (a < 6400)  return o0[((size_t)b * 6400 + a) * 85 + c];
  if (a < 8000)  return o1[((size_t)b * 1600 + (a - 6400)) * 85 + c];
  return o2[((size_t)b * 400 + (a - 8000)) * 85 + c];
}

__device__ __forceinline__ void anchor_geom(int a, float& sx, float& sy, float& st) {
  if (a < 6400)      { st = 8.0f;  sx = (float)(a % 80); sy = (float)(a / 80); }
  else if (a < 8000) { int i = a - 6400; st = 16.0f; sx = (float)(i % 40); sy = (float)(i / 40); }
  else               { int i = a - 8000; st = 32.0f; sx = (float)(i % 20); sy = (float)(i / 20); }
}

// ---------------------------------------------------------------------------
// K1: per (batch, 16-anchor tile): build M (f16) + S in LDS, run WMMA for the
// one-hot class-cost matmul, fuse pairwise IoU / masks, emit cost & iou rows.
// One wave32 per block -> EXEC all ones around v_wmma (required by ISA).
// ---------------------------------------------------------------------------
__global__ __launch_bounds__(32)
void k1_cost(const float* __restrict__ o0, const float* __restrict__ o1,
             const float* __restrict__ o2, const float* __restrict__ tgt,
             float* __restrict__ cost, float* __restrict__ iou,
             float* __restrict__ bbox, int* __restrict__ fg) {
  __shared__ float gtx[G_], gty[G_], gtw[G_], gth[G_];
  __shared__ int   gcls[G_], gval[G_];
  __shared__ _Float16 Mh[96][16];                 // (log1p - logp), K-padded to 96
  __shared__ float S_s[16], sobj[16];
  __shared__ float bbx[16][4], ccx[16], ccy[16], cst[16];
  __shared__ int   fg_s[16];

  const int tid  = threadIdx.x;
  const int b    = blockIdx.y;
  const int base = blockIdx.x * 16;

  // GT boxes / classes / validity
  for (int g = tid; g < G_; g += 32) {
    const float* t = &tgt[((size_t)b * G_ + g) * 5];
    float c0 = t[0], x = t[1], y = t[2], w = t[3], h = t[4];
    gcls[g] = (int)c0;
    gtx[g] = x; gty[g] = y; gtw[g] = w; gth[g] = h;
    gval[g] = (c0 + x + y + w + h) > 0.0f;
  }
  // zero K-pad rows 80..95 of M
  for (int e = tid; e < 16 * 16; e += 32) Mh[80 + (e >> 4)][e & 15] = (_Float16)0.0f;

  // per-anchor decode: bbox, sigmoid(obj), anchor centers
  if (tid < 16) {
    int a = base + tid;
    float sx, sy, st; anchor_geom(a, sx, sy, st);
    float tx = flat_ld(o0, o1, o2, b, a, 0);
    float ty = flat_ld(o0, o1, o2, b, a, 1);
    float tw = flat_ld(o0, o1, o2, b, a, 2);
    float th = flat_ld(o0, o1, o2, b, a, 3);
    float to = flat_ld(o0, o1, o2, b, a, 4);
    float cx = (tx + sx) * st, cy = (ty + sy) * st;
    float pw = expf(tw) * st,  ph = expf(th) * st;
    bbx[tid][0] = cx; bbx[tid][1] = cy; bbx[tid][2] = pw; bbx[tid][3] = ph;
    float* bb = &bbox[((size_t)b * A_ + a) * 4];
    bb[0] = cx; bb[1] = cy; bb[2] = pw; bb[3] = ph;
    sobj[tid] = sigm(to);
    ccx[tid] = (sx + 0.5f) * st; ccy[tid] = (sy + 0.5f) * st; cst[tid] = st;
    S_s[tid] = 0.0f;
  }
  __syncthreads();

  // M[c][i] = log1p(-p) - log(p), S[i] = -sum_c log1p(-p)
  for (int e = tid; e < NC_ * 16; e += 32) {
    int c = e >> 4, i = e & 15;
    float lg  = flat_ld(o0, o1, o2, b, base + i, 5 + c);
    float p   = sqrtf(sigm(lg) * sobj[i]);
    float lp  = fmaxf(logf(p), -100.0f);
    float l1p = fmaxf(log1pf(-p), -100.0f);
    Mh[c][i] = (_Float16)(l1p - lp);
    atomicAdd(&S_s[i], -l1p);
  }
  __syncthreads();

  // fg[a] = any_g (in_box | in_ctr) & gvalid
  if (tid < 16) {
    float cx = ccx[tid], cy = ccy[tid], st = cst[tid];
    int f = 0;
    for (int g = 0; g < G_; ++g) {
      if (!gval[g]) continue;
      float hw = 0.5f * gtw[g], hh = 0.5f * gth[g];
      int ib = (cx - (gtx[g] - hw) > 0.0f) && ((gtx[g] + hw) - cx > 0.0f) &&
               (cy - (gty[g] - hh) > 0.0f) && ((gty[g] + hh) - cy > 0.0f);
      float cr = 2.5f * st;
      int ic = (cx - (gtx[g] - cr) > 0.0f) && ((gtx[g] + cr) - cx > 0.0f) &&
               (cy - (gty[g] - cr) > 0.0f) && ((gty[g] + cr) - cy > 0.0f);
      if (ib || ic) { f = 1; break; }
    }
    fg_s[tid] = f;
    fg[(size_t)b * A_ + base + tid] = f;
  }
  __syncthreads();

  const int lane = tid;
  const int hi   = (lane >= 16) ? 1 : 0;
  const int n    = lane & 15;

  for (int gt4 = 0; gt4 < 4; ++gt4) {
    v8f acc = {0.f, 0.f, 0.f, 0.f, 0.f, 0.f, 0.f, 0.f};
    const int grow = gt4 * 16 + n;                 // A-matrix row M = lane&15
    const int gc   = (grow < G_) ? gcls[grow] : -1;
    for (int kb = 0; kb < 96; kb += 32) {
      v16h av, bv;
#pragma unroll
      for (int j = 0; j < 16; ++j) {
        // 16-bit A 16x32 layout: VGPR0..3 K=0..7 (+8 for hi lanes), VGPR4..7 K=16..23 (+8)
        int ka = kb + ((j < 8) ? 0 : 16) + (hi ? 8 : 0) + (j & 7);
        av[j] = (_Float16)((ka == gc) ? 1.0f : 0.0f);
        // 16-bit B 32x16 layout: same K order as A (K=0..7,16..23; +8 for hi lanes), N=lane&15
        int kbb = kb + ((j < 8) ? 0 : 16) + (hi ? 8 : 0) + (j & 7);
        bv[j] = Mh[kbb][n];
      }
      acc = __builtin_amdgcn_wmma_f32_16x16x32_f16(
          false, av, false, bv, (short)0, acc, false, false);
    }
    // D layout: VGPR r -> M = r (lanes0-15) / 8+r (lanes16-31), N = lane&15
#pragma unroll
    for (int r = 0; r < 8; ++r) {
      int g = gt4 * 16 + r + (hi ? 8 : 0);
      if (g >= G_) continue;
      int a = base + n;
      float px = bbx[n][0], py = bbx[n][1], pw = bbx[n][2], ph = bbx[n][3];
      float gx = gtx[g], gy = gty[g], gw = gtw[g], gh = gth[g];
      float iw = fminf(gx + 0.5f * gw, px + 0.5f * pw) - fmaxf(gx - 0.5f * gw, px - 0.5f * pw);
      float ih = fminf(gy + 0.5f * gh, py + 0.5f * ph) - fmaxf(gy - 0.5f * gh, py - 0.5f * ph);
      iw = fmaxf(iw, 0.0f); ih = fmaxf(ih, 0.0f);
      float inter = iw * ih;
      float vio = inter / (gw * gh + pw * ph - inter + 1e-16f);
      iou[((size_t)b * G_ + g) * A_ + a] = vio;

      float cx = ccx[n], cy = ccy[n], st = cst[n];
      int gv = gval[g];
      float hw = 0.5f * gw, hh = 0.5f * gh;
      int ib = gv && (cx - (gx - hw) > 0.0f) && ((gx + hw) - cx > 0.0f) &&
                     (cy - (gy - hh) > 0.0f) && ((gy + hh) - cy > 0.0f);
      float cr = 2.5f * st;
      int ic = gv && (cx - (gx - cr) > 0.0f) && ((gx + cr) - cx > 0.0f) &&
                     (cy - (gy - cr) > 0.0f) && ((gy + cr) - cy > 0.0f);
      int both = ib && ic;

      float cls_cost = S_s[n] + acc[r];
      float c = 1.0f * cls_cost + 3.0f * (-logf(vio + 1e-8f)) + (both ? 0.0f : 100000.0f);
      if (!fg_s[n]) c = 1e9f;
      if (!gv)      c = 1e15f;
      cost[((size_t)b * GP_ + g) * A_ + a] = c;
    }
  }
}

// ---------------------------------------------------------------------------
// K2: per (b,g): top-10 masked IoU -> dyn_k, then dyn_k rounds of stable
// (value, index) min over the cost row; scatter match flags.
// ---------------------------------------------------------------------------
__global__ __launch_bounds__(256)
void k2_select(const float* __restrict__ tgt, const float* __restrict__ cost,
               const float* __restrict__ iou, const int* __restrict__ fg,
               unsigned char* __restrict__ match) {
  const int g = blockIdx.x, b = blockIdx.y, tid = threadIdx.x;
  __shared__ float rv[256];
  __shared__ int   ri[256];
  __shared__ int   chosen[10];
  __shared__ int   ch2[12];
  __shared__ float topsum;
  __shared__ int   dynk;

  const float* t = &tgt[((size_t)b * G_ + g) * 5];
  if ((t[0] + t[1] + t[2] + t[3] + t[4]) <= 0.0f) return;   // invalid GT: no matches

  const float* irow = &iou[((size_t)b * G_ + g) * A_];
  const float* crow = &cost[((size_t)b * GP_ + g) * A_];
  const int*   frow = &fg[(size_t)b * A_];

  if (tid == 0) topsum = 0.0f;
  __syncthreads();

  // top-10 of iou_c = fg ? iou : 0 (index-excluded rounds)
  for (int k = 0; k < 10; ++k) {
    float bvv = -1.0f; int bii = 0x7fffffff;
    for (int a = tid; a < A_; a += 256) {
      __builtin_prefetch(&irow[a + 2048], 0, 0);
      bool skip = false;
      for (int j = 0; j < k; ++j) if (chosen[j] == a) skip = true;
      if (skip) continue;
      float v = frow[a] ? irow[a] : 0.0f;
      if (v > bvv || (v == bvv && a < bii)) { bvv = v; bii = a; }
    }
    rv[tid] = bvv; ri[tid] = bii; __syncthreads();
    for (int s = 128; s > 0; s >>= 1) {
      if (tid < s) {
        if (rv[tid + s] > rv[tid] || (rv[tid + s] == rv[tid] && ri[tid + s] < ri[tid])) {
          rv[tid] = rv[tid + s]; ri[tid] = ri[tid + s];
        }
      }
      __syncthreads();
    }
    if (tid == 0) { chosen[k] = ri[0]; topsum += fmaxf(rv[0], 0.0f); }
    __syncthreads();
  }
  if (tid == 0) { int dk = (int)topsum; dynk = dk < 1 ? 1 : (dk > 12 ? 12 : dk); }
  __syncthreads();
  const int DK = dynk;

  // dyn_k smallest costs, ties broken by lower index (stable-rank semantics)
  for (int k = 0; k < DK; ++k) {
    float bvv = 3.0e38f; int bii = 0x7fffffff;
    for (int a = tid; a < A_; a += 256) {
      __builtin_prefetch(&crow[a + 2048], 0, 0);
      bool skip = false;
      for (int j = 0; j < k; ++j) if (ch2[j] == a) skip = true;
      if (skip) continue;
      float v = crow[a];
      if (v < bvv || (v == bvv && a < bii)) { bvv = v; bii = a; }
    }
    rv[tid] = bvv; ri[tid] = bii; __syncthreads();
    for (int s = 128; s > 0; s >>= 1) {
      if (tid < s) {
        if (rv[tid + s] < rv[tid] || (rv[tid + s] == rv[tid] && ri[tid + s] < ri[tid])) {
          rv[tid] = rv[tid + s]; ri[tid] = ri[tid + s];
        }
      }
      __syncthreads();
    }
    if (tid == 0) { ch2[k] = ri[0]; match[((size_t)b * G_ + g) * A_ + ri[0]] = 1; }
    __syncthreads();
  }
}

// ---------------------------------------------------------------------------
// K3: per anchor: resolve multi-matches via argmin over cost column; emit
// matched / fgf / pred_iou; reduce num_fg.
// ---------------------------------------------------------------------------
__global__ __launch_bounds__(256)
void k3_resolve(const float* __restrict__ cost, const float* __restrict__ iou,
                const unsigned char* __restrict__ match, int* __restrict__ matched,
                float* __restrict__ fgf, float* __restrict__ piou,
                float* __restrict__ acc) {
  __shared__ float red[256];
  const int idx = blockIdx.x * 256 + threadIdx.x;
  float f = 0.0f;
  if (idx < B_ * A_) {
    const int b = idx / A_, a = idx % A_;
    int cnt = 0, first = 0; bool any = false;
    for (int g = 0; g < G_; ++g) {
      if (match[((size_t)b * G_ + g) * A_ + a]) { ++cnt; if (!any) { first = g; any = true; } }
    }
    if (cnt > 1) {
      int best = 0; float bc = cost[((size_t)b * GP_ + 0) * A_ + a];
      for (int g = 1; g < G_; ++g) {
        float c = cost[((size_t)b * GP_ + g) * A_ + a];
        if (c < bc) { bc = c; best = g; }
      }
      first = best; any = true;
    }
    float pi = any ? iou[((size_t)b * G_ + first) * A_ + a] : 0.0f;
    matched[idx] = first;
    f = any ? 1.0f : 0.0f;
    fgf[idx]  = f;
    piou[idx] = pi;
  }
  red[threadIdx.x] = f; __syncthreads();
  for (int s = 128; s > 0; s >>= 1) {
    if (threadIdx.x < s) red[threadIdx.x] += red[threadIdx.x + s];
    __syncthreads();
  }
  if (threadIdx.x == 0) atomicAdd(&acc[0], red[0]);
}

// ---------------------------------------------------------------------------
// K4: per-anchor loss terms (CIoU, L1, BCE-obj, soft one-hot CE), block-reduced
// into global accumulators.
// ---------------------------------------------------------------------------
__global__ __launch_bounds__(256)
void k4_loss(const float* __restrict__ o0, const float* __restrict__ o1,
             const float* __restrict__ o2, const float* __restrict__ tgt,
             const float* __restrict__ bbox, const int* __restrict__ matched,
             const float* __restrict__ fgf, const float* __restrict__ piou,
             float* __restrict__ acc) {
  __shared__ float r0[256], r1[256], r2[256], r3[256];
  const int idx = blockIdx.x * 256 + threadIdx.x;
  float liou = 0.0f, ll1 = 0.0f, lobj = 0.0f, lcls = 0.0f;
  if (idx < B_ * A_) {
    const int b = idx / A_, a = idx % A_;
    const float fgv = fgf[idx];
    float x = flat_ld(o0, o1, o2, b, a, 4);
    lobj = fmaxf(x, 0.0f) - x * fgv + log1pf(expf(-fabsf(x)));
    if (fgv > 0.0f) {
      const int m = matched[idx];
      const float* t = &tgt[((size_t)b * G_ + m) * 5];
      const int mcls = (int)t[0];
      float gx = t[1], gy = t[2], gw = t[3], gh = t[4];
      float sx, sy, st; anchor_geom(a, sx, sy, st);
      // L1 on raw regression channels
      float l1x = gx / st - sx, l1y = gy / st - sy;
      float l1w = logf(gw / st + 1e-8f), l1h = logf(gh / st + 1e-8f);
      float p0 = flat_ld(o0, o1, o2, b, a, 0);
      float p1 = flat_ld(o0, o1, o2, b, a, 1);
      float p2 = flat_ld(o0, o1, o2, b, a, 2);
      float p3 = flat_ld(o0, o1, o2, b, a, 3);
      ll1 = fabsf(p0 - l1x) + fabsf(p1 - l1y) + fabsf(p2 - l1w) + fabsf(p3 - l1h);
      // CIoU
      const float* bb = &bbox[(size_t)idx * 4];
      float px = bb[0], py = bb[1], pw = bb[2], ph = bb[3];
      float px1 = px - pw * 0.5f, px2 = px + pw * 0.5f;
      float py1 = py - ph * 0.5f, py2 = py + ph * 0.5f;
      float tx1 = gx - gw * 0.5f, tx2 = gx + gw * 0.5f;
      float ty1 = gy - gh * 0.5f, ty2 = gy + gh * 0.5f;
      float iw = fmaxf(fminf(px2, tx2) - fmaxf(px1, tx1), 0.0f);
      float ih = fmaxf(fminf(py2, ty2) - fmaxf(py1, ty1), 0.0f);
      float inter = iw * ih;
      float iouv  = inter / (pw * ph + gw * gh - inter + 1e-7f);
      float cw = fmaxf(px2, tx2) - fminf(px1, tx1);
      float ch = fmaxf(py2, ty2) - fminf(py1, ty1);
      float c2 = cw * cw + ch * ch + 1e-7f;
      float dx = tx1 + tx2 - px1 - px2, dy = ty1 + ty2 - py1 - py2;
      float rho2 = (dx * dx + dy * dy) * 0.25f;
      float dv = atanf(gw / gh) - atanf(pw / ph);
      float v  = (4.0f / (PI_F * PI_F)) * dv * dv;
      float alpha = v / (v - iouv + 1.0f + 1e-7f);
      liou = 1.0f - (iouv - (rho2 / c2 + v * alpha));
      // soft one-hot cross entropy: -pred_iou * log_softmax(cls)[mcls]
      float mx = -3.0e38f;
      for (int c = 0; c < NC_; ++c) mx = fmaxf(mx, flat_ld(o0, o1, o2, b, a, 5 + c));
      float se = 0.0f;
      for (int c = 0; c < NC_; ++c) se += expf(flat_ld(o0, o1, o2, b, a, 5 + c) - mx);
      float lsm = flat_ld(o0, o1, o2, b, a, 5 + mcls) - mx - logf(se);
      lcls = -piou[idx] * lsm;
    }
  }
  r0[threadIdx.x] = liou; r1[threadIdx.x] = ll1;
  r2[threadIdx.x] = lobj; r3[threadIdx.x] = lcls;
  __syncthreads();
  for (int s = 128; s > 0; s >>= 1) {
    if (threadIdx.x < s) {
      r0[threadIdx.x] += r0[threadIdx.x + s];
      r1[threadIdx.x] += r1[threadIdx.x + s];
      r2[threadIdx.x] += r2[threadIdx.x + s];
      r3[threadIdx.x] += r3[threadIdx.x + s];
    }
    __syncthreads();
  }
  if (threadIdx.x == 0) {
    atomicAdd(&acc[1], r0[0]);
    atomicAdd(&acc[2], r1[0]);
    atomicAdd(&acc[3], r2[0]);
    atomicAdd(&acc[4], r3[0]);
  }
}

__global__ void k5_final(const float* __restrict__ acc, float* __restrict__ out) {
  if (threadIdx.x == 0 && blockIdx.x == 0) {
    float nf = fmaxf(acc[0], 1.0f);
    out[0] = 5.0f * acc[1] / nf;               // REG_W * loss_iou
    out[1] = acc[2] / (nf * 4.0f);             // loss_l1
    out[2] = acc[3] / (float)(B_ * A_);        // loss_obj (mean)
    out[3] = acc[4] / nf;                      // loss_cls
  }
}

// ---------------------------------------------------------------------------
extern "C" void kernel_launch(void* const* d_in, const int* in_sizes, int n_in,
                              void* d_out, int out_size, void* d_ws, size_t ws_size,
                              hipStream_t stream) {
  (void)in_sizes; (void)n_in; (void)out_size; (void)ws_size;
  const float* o0  = (const float*)d_in[0];
  const float* o1  = (const float*)d_in[1];
  const float* o2  = (const float*)d_in[2];
  const float* tgt = (const float*)d_in[3];
  float* out = (float*)d_out;

  char* ws = (char*)d_ws;
  size_t off = 0;
  auto carve = [&](size_t bytes) -> void* {
    void* p = ws + off;
    off += (bytes + 511) & ~(size_t)511;
    return p;
  };
  float*         cost    = (float*)carve(sizeof(float) * (size_t)B_ * GP_ * A_);
  float*         iou     = (float*)carve(sizeof(float) * (size_t)B_ * G_ * A_);
  float*         bbox    = (float*)carve(sizeof(float) * (size_t)B_ * A_ * 4);
  int*           fg      = (int*)carve(sizeof(int) * (size_t)B_ * A_);
  unsigned char* match   = (unsigned char*)carve((size_t)B_ * G_ * A_);
  int*           matched = (int*)carve(sizeof(int) * (size_t)B_ * A_);
  float*         fgfv    = (float*)carve(sizeof(float) * (size_t)B_ * A_);
  float*         piou    = (float*)carve(sizeof(float) * (size_t)B_ * A_);
  float*         acc     = (float*)carve(sizeof(float) * 8);

  hipMemsetAsync(match, 0, (size_t)B_ * G_ * A_, stream);
  hipMemsetAsync(acc, 0, sizeof(float) * 8, stream);

  dim3 g1(A_ / 16, B_);
  k1_cost<<<g1, 32, 0, stream>>>(o0, o1, o2, tgt, cost, iou, bbox, fg);

  dim3 g2(G_, B_);
  k2_select<<<g2, 256, 0, stream>>>(tgt, cost, iou, fg, match);

  const int nBlk = (B_ * A_ + 255) / 256;
  k3_resolve<<<nBlk, 256, 0, stream>>>(cost, iou, match, matched, fgfv, piou, acc);
  k4_loss<<<nBlk, 256, 0, stream>>>(o0, o1, o2, tgt, bbox, matched, fgfv, piou, acc);
  k5_final<<<1, 32, 0, stream>>>(acc, out);
}
